// RoughScorer_45767171506490
// MI455X (gfx1250) — matrix operand order repaired and checked by reading, not compiled, not run on hardware
//
#include <hip/hip_runtime.h>

#define N_MENT 8192
#define FEAT   1024
#define K_TOP  50

typedef __attribute__((ext_vector_type(16))) __bf16 v16bf;
typedef __attribute__((ext_vector_type(8)))  __bf16 v8bf;
typedef __attribute__((ext_vector_type(8)))  float  v8f;

union V16U { v16bf v; v8bf h[2]; };

#define NEG_INF (-__builtin_inff())

__device__ __forceinline__ unsigned short f2bf(float f) {
    unsigned u = __float_as_uint(f);
    u += 0x7FFFu + ((u >> 16) & 1u);          // round-to-nearest-even
    return (unsigned short)(u >> 16);
}

// ---------------------------------------------------------------- convert
__global__ void cvt_f32_bf16(const float* __restrict__ in,
                             unsigned short* __restrict__ out, int n) {
    int i = blockIdx.x * blockDim.x + threadIdx.x;
    int stride = gridDim.x * blockDim.x;
    for (; i < n; i += stride) out[i] = f2bf(in[i]);
}

// ---------------------------------------------------------------- GEMM1: proj = mentions @ W^T + b  (bf16 out)
__global__ __launch_bounds__(256) void proj_gemm(
    const unsigned short* __restrict__ M,   // mentions bf16 [8192][1024]
    const unsigned short* __restrict__ Wb,  // W bf16 [1024][1024]
    const float* __restrict__ bias,
    unsigned short* __restrict__ P)         // proj bf16 [8192][1024]
{
    const int wave = threadIdx.x >> 5;
    const int lane = threadIdx.x & 31;
    const int band = blockIdx.x >> 3;       // 512 row bands of 16
    const int cg   = blockIdx.x & 7;        // 8 col groups of 128
    const int rowBase = band * 16;
    const int nBase   = cg * 128 + wave * 16;

    const int mrow = lane & 15;
    const int kA   = (lane < 16) ? 0 : 8;   // A: 16-bit layout, K split per lane half
    const int kB   = (lane < 16) ? 0 : 8;   // B: same K split as A
    const int ncol = lane & 15;

    const unsigned short* aRow = M  + (size_t)(rowBase + mrow) * FEAT;
    const unsigned short* bRow = Wb + (size_t)(nBase + ncol) * FEAT;  // B[k][n] = W[n][k]

    v8f acc = {};
    for (int k0 = 0; k0 < FEAT; k0 += 32) {
        V16U a;
        a.h[0] = *(const v8bf*)(aRow + k0 + kA);        // K = kA .. kA+7
        a.h[1] = *(const v8bf*)(aRow + k0 + kA + 16);   // K = kA+16 .. kA+23
        V16U b;
        b.h[0] = *(const v8bf*)(bRow + k0 + kB);        // K = kB .. kB+7
        b.h[1] = *(const v8bf*)(bRow + k0 + kB + 16);   // K = kB+16 .. kB+23
        acc = __builtin_amdgcn_wmma_f32_16x16x32_bf16(
                  false, a.v, false, b.v, (short)0, acc, false, false);
    }

    const int col  = nBase + ncol;
    const float bb = bias[col];
    const int mOff = (lane < 16) ? 0 : 8;
    for (int v = 0; v < 8; ++v) {
        int row = rowBase + mOff + v;
        P[(size_t)row * FEAT + col] = f2bf(acc[v] + bb);
    }
}

// ---------------------------------------------------------------- GEMM2 + fused masked top-k
__global__ __launch_bounds__(256) void score_topk(
    const unsigned short* __restrict__ M,   // mentions bf16
    const unsigned short* __restrict__ P,   // proj bf16
    float* __restrict__ outScore, int* __restrict__ outIdx)
{
    __shared__ __align__(16) unsigned short ldsA[16 * FEAT];   // 32 KB band of proj
    __shared__ float ldsScore[16][128];                        // 8 KB
    __shared__ float candVal[16][128];                         // 8 KB
    __shared__ int   candIdx[16][128];                         // 8 KB
    __shared__ int   candCnt[16];
    __shared__ float heapVal[16][K_TOP];                       // 3.2 KB
    __shared__ int   heapIdx[16][K_TOP];                       // 3.2 KB
    __shared__ float heapMin[16];

    const int tid  = threadIdx.x;
    const int wave = tid >> 5;
    const int lane = tid & 31;
    const int band = (int)gridDim.x - 1 - (int)blockIdx.x;     // longest bands first
    const int rowBase = band * 16;

    if (tid < 16) { candCnt[tid] = 0; heapMin[tid] = NEG_INF; }
    for (int i = tid; i < 16 * K_TOP; i += 256) {
        heapVal[i / K_TOP][i % K_TOP] = NEG_INF;
        heapIdx[i / K_TOP][i % K_TOP] = 0;
    }

    // stage proj band -> LDS with CDNA5 async copies (ASYNCcnt-tracked, no VGPR round-trip)
    {
        const unsigned long long gbase =
            (unsigned long long)(uintptr_t)(const void*)(P + (size_t)rowBase * FEAT);
        const unsigned ldsBase = (unsigned)(uintptr_t)(void*)ldsA; // low 32 bits = LDS offset
        for (int i = tid; i < (16 * FEAT * 2) / 16; i += 256) {    // 2048 x 16B chunks
            unsigned long long g = gbase + (unsigned long long)i * 16ull;
            unsigned l = ldsBase + (unsigned)i * 16u;
            asm volatile("global_load_async_to_lds_b128 %0, %1, off"
                         :: "v"(l), "v"(g) : "memory");
        }
        asm volatile("s_wait_asynccnt 0" ::: "memory");
    }
    __syncthreads();

    const int mrow = lane & 15;
    const int kA   = (lane < 16) ? 0 : 8;
    const int kB   = (lane < 16) ? 0 : 8;
    const int mOff = (lane < 16) ? 0 : 8;
    const unsigned short* aRow = ldsA + mrow * FEAT;

    const int nIter = (rowBase + 15 + 127) >> 7;               // cols j <= rowBase+14 needed
    for (int it = 0; it < nIter; ++it) {
        const int colBase = it * 128;
        const int waveCol = colBase + wave * 16;
        if (waveCol < rowBase + 15) {                          // wave-uniform skip past mask
            const int gcol = waveCol + (lane & 15);
            const unsigned short* bRow = M + (size_t)gcol * FEAT;  // B[k][n] = mentions[n][k]
            if (it + 1 < nIter)                                // warm L2->L0 path for next tile
                __builtin_prefetch(bRow + (size_t)128 * FEAT, 0, 0);
            v8f acc = {};
            for (int k0 = 0; k0 < FEAT; k0 += 32) {
                V16U a;
                a.h[0] = *(const v8bf*)(aRow + k0 + kA);
                a.h[1] = *(const v8bf*)(aRow + k0 + kA + 16);
                V16U b;
                b.h[0] = *(const v8bf*)(bRow + k0 + kB);
                b.h[1] = *(const v8bf*)(bRow + k0 + kB + 16);
                acc = __builtin_amdgcn_wmma_f32_16x16x32_bf16(
                          false, a.v, false, b.v, (short)0, acc, false, false);
            }
            const int c = wave * 16 + (lane & 15);
            for (int v = 0; v < 8; ++v) ldsScore[mOff + v][c] = acc[v];
        }
        __syncthreads();

        // parallel mask + threshold filter (stale heapMin is conservative: it only grows)
        {
            const int r  = tid >> 4;
            const int c0 = (tid & 15) * 8;
            const int grow = rowBase + r;
            const float hmin = heapMin[r];
            for (int i = 0; i < 8; ++i) {
                const int gcol = colBase + c0 + i;
                if (gcol < grow) {
                    const float s = ldsScore[r][c0 + i];
                    if (s > hmin) {
                        int slot = atomicAdd(&candCnt[r], 1); // <=128 per row per iter
                        candVal[r][slot] = s;
                        candIdx[r][slot] = gcol;
                    }
                }
            }
        }
        __syncthreads();

        // serial drain into per-row 50-entry min-heap (expected few survivors)
        if (tid < 16) {
            const int r = tid;
            const int cnt = candCnt[r];
            for (int i = 0; i < cnt; ++i) {
                const float cv = candVal[r][i];
                if (cv > heapVal[r][0]) {
                    heapVal[r][0] = cv; heapIdx[r][0] = candIdx[r][i];
                    int p = 0;
                    for (;;) {
                        int l = 2 * p + 1, q = 2 * p + 2, s = p;
                        if (l < K_TOP && heapVal[r][l] < heapVal[r][s]) s = l;
                        if (q < K_TOP && heapVal[r][q] < heapVal[r][s]) s = q;
                        if (s == p) break;
                        float tv = heapVal[r][p]; heapVal[r][p] = heapVal[r][s]; heapVal[r][s] = tv;
                        int   ti = heapIdx[r][p]; heapIdx[r][p] = heapIdx[r][s]; heapIdx[r][s] = ti;
                        p = s;
                    }
                }
            }
            heapMin[r] = heapVal[r][0];
            candCnt[r] = 0;
        }
        __syncthreads();
    }

    // sort descending + jax-style -inf/index fill, write out
    if (tid < 16) {
        const int r = tid;
        const int grow = rowBase + r;
        for (int i = 0; i < K_TOP - 1; ++i) {
            int best = i;
            for (int j = i + 1; j < K_TOP; ++j) {
                float vj = heapVal[r][j], vb = heapVal[r][best];
                if (vj > vb || (vj == vb && heapIdx[r][j] < heapIdx[r][best])) best = j;
            }
            if (best != i) {
                float tv = heapVal[r][i]; heapVal[r][i] = heapVal[r][best]; heapVal[r][best] = tv;
                int   ti = heapIdx[r][i]; heapIdx[r][i] = heapIdx[r][best]; heapIdx[r][best] = ti;
            }
        }
        int mcnt = 0;
        while (mcnt < K_TOP && heapVal[r][mcnt] > NEG_INF) ++mcnt;
        for (int s = 0; s < K_TOP; ++s) {
            float v; int id;
            if (s < mcnt) { v = heapVal[r][s]; id = heapIdx[r][s]; }
            else          { v = NEG_INF;       id = grow + (s - mcnt); } // lowest -inf indices
            outScore[(size_t)grow * K_TOP + s] = v;
            outIdx  [(size_t)grow * K_TOP + s] = id;
        }
    }
}

// ---------------------------------------------------------------- launcher
extern "C" void kernel_launch(void* const* d_in, const int* in_sizes, int n_in,
                              void* d_out, int out_size, void* d_ws, size_t ws_size,
                              hipStream_t stream) {
    (void)in_sizes; (void)n_in; (void)out_size; (void)ws_size;
    const float* mentions = (const float*)d_in[0];
    const float* W        = (const float*)d_in[1];
    const float* bias     = (const float*)d_in[2];
    // d_in[3] = k (== 50, baked in as K_TOP)

    char* ws = (char*)d_ws;
    unsigned short* mBF = (unsigned short*)ws;                                   // 16 MB
    unsigned short* wBF = (unsigned short*)(ws + (size_t)N_MENT * FEAT * 2);     //  2 MB
    unsigned short* pBF = (unsigned short*)(ws + (size_t)N_MENT * FEAT * 2
                                               + (size_t)FEAT * FEAT * 2);       // 16 MB

    cvt_f32_bf16<<<2048, 256, 0, stream>>>(mentions, mBF, N_MENT * FEAT);
    cvt_f32_bf16<<<512,  256, 0, stream>>>(W,        wBF, FEAT * FEAT);

    proj_gemm<<<(N_MENT / 16) * (FEAT / 128), 256, 0, stream>>>(mBF, wBF, bias, pBF);

    float* outScore = (float*)d_out;
    int*   outIdx   = (int*)((float*)d_out + (size_t)N_MENT * K_TOP);
    score_topk<<<N_MENT / 16, 256, 0, stream>>>(mBF, pBF, outScore, outIdx);
}
